// Token_QK_Attention_39367670235566
// MI455X (gfx1250) — matrix operand order repaired and checked by reading, not compiled, not run on hardware
//
#include <hip/hip_runtime.h>

#define T_ 4
#define B_ 16
#define C_ 384
#define N_ 1024
#define HEADS_ 8
#define CH_ 48

typedef __attribute__((ext_vector_type(16))) _Float16 v16h;
typedef __attribute__((ext_vector_type(8)))  _Float16 v8h;
typedef __attribute__((ext_vector_type(4)))  _Float16 v4h;
typedef __attribute__((ext_vector_type(8)))  float    v8f;

union V16 { v16h v; v8h h[2]; };

__device__ __forceinline__ v8f wmma_f16(v16h a, v16h b, v8f c) {
  // D = A(16x32 f16) * B(32x16 f16) + C(16x16 f32)
  return __builtin_amdgcn_wmma_f32_16x16x32_f16(false, a, false, b, (short)0, c, false, false);
}

// ---------------------------------------------------------------------------
// prep: convert the four 384x384 weight matrices to fp16, and fold BN (and the
// p-projection bias) into per-channel scale/shift:  y_bn = y*scl[d] + sft[d]
// ---------------------------------------------------------------------------
__global__ void prep_kernel(const float* __restrict__ qw, const float* __restrict__ kw,
                            const float* __restrict__ vw, const float* __restrict__ pw,
                            const float* __restrict__ qg, const float* __restrict__ qb,
                            const float* __restrict__ qm, const float* __restrict__ qv,
                            const float* __restrict__ kg, const float* __restrict__ kb,
                            const float* __restrict__ km, const float* __restrict__ kv,
                            const float* __restrict__ vg, const float* __restrict__ vb,
                            const float* __restrict__ vm, const float* __restrict__ vv,
                            const float* __restrict__ pg, const float* __restrict__ pb,
                            const float* __restrict__ pm, const float* __restrict__ pv,
                            const float* __restrict__ pbias,
                            _Float16* __restrict__ wh, float* __restrict__ scl,
                            float* __restrict__ sft) {
  const int idx = blockIdx.x * 256 + threadIdx.x;
  if (idx < 4 * C_ * C_) {
    const int br = idx / (C_ * C_);
    const int i  = idx % (C_ * C_);
    const float* w = (br == 0) ? qw : (br == 1) ? kw : (br == 2) ? vw : pw;
    wh[idx] = (_Float16)w[i];
  }
  if (idx < 4 * C_) {
    const int br = idx / C_, c = idx % C_;
    const float *g, *bt, *mn, *vr;
    switch (br) {
      case 0:  g = qg; bt = qb; mn = qm; vr = qv; break;
      case 1:  g = kg; bt = kb; mn = km; vr = kv; break;
      case 2:  g = vg; bt = vb; mn = vm; vr = vv; break;
      default: g = pg; bt = pb; mn = pm; vr = pv; break;
    }
    const float s = g[c] * rsqrtf(vr[c] + 1e-5f);
    float f = bt[c] - mn[c] * s;
    if (br == 3) f += pbias[c] * s;   // p-projection: bias added before BN
    scl[idx] = s;
    sft[idx] = f;
  }
}

// ---------------------------------------------------------------------------
// Input LIF with LDS transpose: x (T,B,C,N) fp32 -> spikes xs (B,T,N,C) fp16.
// TAU=2 => h=(v+x)/2 ; spike = h>=1 ; hard reset.
// Block tile: 64c x 64n for one b, all 4 t. Reads are n-coalesced fp32;
// spikes staged in LDS (row stride padded to 68 halves = 34 dwords:
// conflict-free across lanes on 64 banks, 8B-aligned rows), then written back
// as 128B-contiguous c-runs per thread.
// ---------------------------------------------------------------------------
__global__ __launch_bounds__(256) void lif_in_kernel(const float* __restrict__ x,
                                                     _Float16* __restrict__ xs) {
  __shared__ _Float16 lds[T_ * 64 * 68];
  const int tid = threadIdx.x;
  const int nTile = blockIdx.x;       // N/64 = 16
  const int cTile = blockIdx.y;       // C/64 = 6
  const int b     = blockIdx.z;
  const int nl = tid & 63;
  const int c0 = tid >> 6;            // 0..3
#pragma unroll
  for (int s = 0; s < 16; ++s) {
    const int cl = s * 4 + c0;
    const int c  = cTile * 64 + cl;
    const int n  = nTile * 64 + nl;
    float vmem = 0.f;
#pragma unroll
    for (int t = 0; t < T_; ++t) {
      const float xv = x[(((size_t)t * B_ + b) * C_ + c) * N_ + n];
      const float h  = 0.5f * (vmem + xv);
      const bool fire = (h >= 1.0f);
      vmem = fire ? 0.f : h;
      lds[(t * 64 + nl) * 68 + cl] = (_Float16)(fire ? 1.f : 0.f);
    }
  }
  __syncthreads();
  // write-back: one (t, n) row of 64 contiguous channels per thread
  const int t  = tid >> 6;
  const int nw = tid & 63;
  const size_t gbase =
      (((size_t)b * T_ + t) * N_ + (size_t)nTile * 64 + nw) * C_ + (size_t)cTile * 64;
  const _Float16* row = lds + (t * 64 + nw) * 68;
#pragma unroll
  for (int j = 0; j < 16; ++j)
    *(v4h*)(xs + gbase + j * 4) = *(const v4h*)(row + j * 4);
}

// ---------------------------------------------------------------------------
// Fused GEMM + BN (+ LIF) kernel, high-intensity tiling.
//   Y[t] = Wh (C x C, fp16) * X[b,t] (C x N, fp16 spikes, channel-last)
// Wave tile: 64 d (4 m-tiles) x 16 n x 4 t -> 16 WMMA accumulators (128 VGPR).
// Workgroup = 8 waves side by side in n -> WG tile 64d x 128n x 4t; all waves
// share the same A (weight) fragments via L1.
// Per k-step: 8 A-loads + 8 B-loads feed 16 WMMAs (1.0 b128 per WMMA; each
// B fragment is reused by 4 WMMAs, each A fragment by 4 t).
// Per-lane accumulator slots hold the same (d,n) for every t, so the LIF
// recurrence over t runs entirely in registers.
// FINAL=false: blockIdx.z = branch*16 + b (q/k/v fused in one launch for L2
//              reuse of X); epilogue = BN + LIF(vth=1); spikes packed into
//              v8h (8 consecutive d per lane) -> one global_store_b128 each.
// FINAL=true : blockIdx.z = b; epilogue = bias+BN, fp32 out (T,B,C,N),
//              stores lane-coalesced along n.
// ---------------------------------------------------------------------------
template <bool FINAL>
__global__ __launch_bounds__(256, 4) void gemm_kernel(
    const _Float16* __restrict__ X, const _Float16* __restrict__ Wh0,
    const float* __restrict__ scl0, const float* __restrict__ sft0,
    _Float16* __restrict__ Sout0, float* __restrict__ Fout) {
  const int lane = threadIdx.x & 31;
  const int w    = threadIdx.x >> 5;
  const int l16  = lane & 15;
  const int hi   = lane >> 4;          // 0 = lanes 0-15, 1 = lanes 16-31
  const int br   = FINAL ? 0 : (blockIdx.z >> 4);   // branch: q/k/v
  const int b    = FINAL ? blockIdx.z : (blockIdx.z & 15);
  const int dBase = blockIdx.y * 64;                   // WG covers 64 d
  const int nCol  = blockIdx.x * 128 + w * 16 + l16;   // this lane's column n

  const size_t xsE = (size_t)B_ * T_ * N_ * C_;
  const _Float16* Wh = Wh0 + (size_t)br * C_ * C_;
  const float* scl = scl0 + br * C_;
  const float* sft = sft0 + br * C_;
  _Float16* Sout = Sout0 + (size_t)br * xsE;

  v8f acc[T_][4] = {};

  // A operand (weights): row M = l16 + mt*16; K split per ISA:
  // lo lanes: K = e (+8 for e>=8); hi lanes: +8.
  const _Float16* aRow = Wh + (size_t)(dBase + l16) * C_;
  // B operand (spikes): column N = l16; K split per ISA, same as A.
  // Single base pointer; t becomes an immediate offset (t * N*C*2 bytes).
  const _Float16* bBase = X + (((size_t)b * T_) * N_ + nCol) * C_;

  constexpr int KSTEPS = C_ / 32;      // 12
#pragma unroll
  for (int it = 0; it < KSTEPS; ++it) {
    const int k0 = it * 32;
    V16 aF[4], bF[4];
#pragma unroll
    for (int mt = 0; mt < 4; ++mt) {
      const _Float16* ar = aRow + (size_t)mt * 16 * C_;
      aF[mt].h[0] = *(const v8h*)(ar + k0 + hi * 8);
      aF[mt].h[1] = *(const v8h*)(ar + k0 + 16 + hi * 8);
    }
#pragma unroll
    for (int t = 0; t < T_; ++t) {
      const _Float16* brow = bBase + (size_t)t * N_ * C_;
      bF[t].h[0] = *(const v8h*)(brow + k0 + hi * 8);
      bF[t].h[1] = *(const v8h*)(brow + k0 + 16 + hi * 8);
    }
#pragma unroll
    for (int t = 0; t < T_; ++t)
#pragma unroll
      for (int mt = 0; mt < 4; ++mt)
        acc[t][mt] = wmma_f16(aF[mt].v, bF[t].v, acc[t][mt]);
  }

  // Epilogue, one m-tile at a time to bound live registers.
  // C/D layout: VGPR r, lane -> M = r + hi*8, N = l16.
#pragma unroll
  for (int mt = 0; mt < 4; ++mt) {
    float sv[8], fv[8];
#pragma unroll
    for (int r = 0; r < 8; ++r) {
      const int d = dBase + mt * 16 + hi * 8 + r;
      sv[r] = scl[d];
      fv[r] = sft[d];
    }
    if (!FINAL) {
      float vmem[8] = {};
#pragma unroll
      for (int t = 0; t < T_; ++t) {
        v8h sp;
#pragma unroll
        for (int r = 0; r < 8; ++r) {
          const float y = acc[t][mt][r] * sv[r] + fv[r];   // BatchNorm
          const float h = 0.5f * (vmem[r] + y);            // LIF, TAU=2
          const bool fire = (h >= 1.0f);
          vmem[r] = fire ? 0.f : h;                        // hard reset
          sp[r] = (_Float16)(fire ? 1.f : 0.f);
        }
        // 8 consecutive d per lane -> one aligned 16B store
        *(v8h*)(Sout + (((size_t)b * T_ + t) * N_ + nCol) * C_ + dBase +
                mt * 16 + hi * 8) = sp;
      }
    } else {
#pragma unroll
      for (int r = 0; r < 8; ++r) {
        const int d = dBase + mt * 16 + hi * 8 + r;
#pragma unroll
        for (int t = 0; t < T_; ++t) {
          const float y = acc[t][mt][r] * sv[r] + fv[r];   // bias+BN
          Fout[(((size_t)t * B_ + b) * C_ + d) * N_ + nCol] = y;
        }
      }
    }
  }
}

// ---------------------------------------------------------------------------
// attn = LIF_{0.5}( sum_ch k*v ) per (t,b,head,n); then q *= attn in place.
// Spikes are {0,1}, so q*attn is "keep or zero".
// ---------------------------------------------------------------------------
__global__ void attn_kernel(const _Float16* __restrict__ ks,
                            const _Float16* __restrict__ vs,
                            _Float16* __restrict__ qs) {
  const int idx = blockIdx.x * 256 + threadIdx.x;       // over B*HEADS*N
  const int n = idx % N_;
  const int h = (idx / N_) % HEADS_;
  const int b = idx / (N_ * HEADS_);
  float vmem = 0.f;
#pragma unroll
  for (int t = 0; t < T_; ++t) {
    const size_t base = (((size_t)b * T_ + t) * N_ + n) * C_ + (size_t)h * CH_;
    float s = 0.f;
#pragma unroll
    for (int j = 0; j < CH_ / 8; ++j) {
      const v8h kk = *(const v8h*)(ks + base + j * 8);
      const v8h vv = *(const v8h*)(vs + base + j * 8);
#pragma unroll
      for (int e = 0; e < 8; ++e) s += (float)kk[e] * (float)vv[e];
    }
    const float hc = 0.5f * (vmem + s);
    const bool fire = (hc >= 0.5f);                     // vth = 0.5
    vmem = fire ? 0.f : hc;
    if (!fire) {                                        // attn==0 -> zero q block
      const v8h z = {};
#pragma unroll
      for (int j = 0; j < CH_ / 8; ++j) *(v8h*)(qs + base + j * 8) = z;
    }
  }
}

// ---------------------------------------------------------------------------
extern "C" void kernel_launch(void* const* d_in, const int* in_sizes, int n_in,
                              void* d_out, int out_size, void* d_ws, size_t ws_size,
                              hipStream_t stream) {
  (void)in_sizes; (void)n_in; (void)out_size; (void)ws_size;
  const float* x  = (const float*)d_in[0];
  const float* qw = (const float*)d_in[1];
  const float* qg = (const float*)d_in[2],  *qb = (const float*)d_in[3];
  const float* qm = (const float*)d_in[4],  *qv = (const float*)d_in[5];
  const float* kw = (const float*)d_in[6];
  const float* kg = (const float*)d_in[7],  *kb = (const float*)d_in[8];
  const float* km = (const float*)d_in[9],  *kv = (const float*)d_in[10];
  const float* vw = (const float*)d_in[11];
  const float* vg = (const float*)d_in[12], *vb = (const float*)d_in[13];
  const float* vm = (const float*)d_in[14], *vv = (const float*)d_in[15];
  const float* pw = (const float*)d_in[16];
  const float* pg = (const float*)d_in[17], *pb = (const float*)d_in[18];
  const float* pm = (const float*)d_in[19], *pv = (const float*)d_in[20];
  const float* pbias = (const float*)d_in[21];

  // Workspace layout (fp16 spike tensors in (B,T,N,C) layout).
  // q/k/v spike buffers are contiguous so the fused branch-GEMM can offset by
  // branch index.
  const size_t xsE = (size_t)B_ * T_ * N_ * C_;     // 25,165,824 elements
  _Float16* xs  = (_Float16*)d_ws;
  _Float16* qsp = xs  + xsE;                        // branch 0
  _Float16* ksp = qsp + xsE;                        // branch 1
  _Float16* vsp = ksp + xsE;                        // branch 2
  _Float16* wh  = vsp + xsE;                        // 4 * C*C fp16 weights
  float* scl = (float*)(wh + (size_t)4 * C_ * C_);  // 4*C
  float* sft = scl + 4 * C_;                        // 4*C

  prep_kernel<<<(4 * C_ * C_ + 255) / 256, 256, 0, stream>>>(
      qw, kw, vw, pw, qg, qb, qm, qv, kg, kb, km, kv,
      vg, vb, vm, vv, pg, pb, pm, pv, pbias, wh, scl, sft);

  lif_in_kernel<<<dim3(N_ / 64, C_ / 64, B_), 256, 0, stream>>>(x, xs);

  // q,k,v channel-mix GEMMs fused into one launch: z = branch*16 + b
  gemm_kernel<false><<<dim3(N_ / 128, C_ / 64, 3 * B_), 256, 0, stream>>>(
      xs, wh, scl, sft, qsp, nullptr);

  attn_kernel<<<(B_ * HEADS_ * N_) / 256, 256, 0, stream>>>(ksp, vsp, qsp);

  gemm_kernel<true><<<dim3(N_ / 128, C_ / 64, B_), 256, 0, stream>>>(
      qsp, wh + (size_t)3 * C_ * C_, scl + 3 * C_, sft + 3 * C_,
      nullptr, (float*)d_out);
}
